// _CrossAttention_39599598469644
// MI455X (gfx1250) — matrix operand order repaired and checked by reading, not compiled, not run on hardware
//
#include <hip/hip_runtime.h>

// ---------------------------------------------------------------------------
// Cross-attention for gfx1250 (MI455X), bf16 WMMA pipeline + async-to-LDS.
//   B=4, L=4096, R=512, DIM=1024, H=16, DH=64, INNER=1024, SCALE=1/8
// ---------------------------------------------------------------------------

typedef __attribute__((ext_vector_type(8)))  float  v8f;
typedef __attribute__((ext_vector_type(16))) __bf16 v16bf;
typedef __attribute__((ext_vector_type(8)))  __bf16 v8bf;
typedef __attribute__((ext_vector_type(4)))  __bf16 v4bf;

#define ATTN_SCALE 0.125f

__device__ inline v8f wmma_bf16(v16bf a, v16bf b, v8f c) {
  // D = A(16x32 bf16) * B(32x16 bf16) + C(16x16 f32)
  return __builtin_amdgcn_wmma_f32_16x16x32_bf16(
      /*neg_a=*/false, a, /*neg_b=*/false, b,
      /*c_mod=*/(short)0, c, /*reuse_a=*/false, /*reuse_b=*/false);
}

// Assemble a 16-element bf16 fragment from two contiguous 8-element LDS reads.
__device__ inline v16bf frag2(const __bf16* p0, const __bf16* p1) {
  union { v16bf v; v8bf h[2]; } u;
  u.h[0] = *reinterpret_cast<const v8bf*>(p0);
  u.h[1] = *reinterpret_cast<const v8bf*>(p1);
  return u.v;
}

// ---------------------------------------------------------------------------
// CDNA5 async global->LDS copy (16B per lane), tracked by ASYNCcnt.
// ---------------------------------------------------------------------------
__device__ inline void async_copy_b128(void* lds_dst, const void* gsrc) {
  // LDS byte offset: generic -> AS3 -> integer (AS3 ptr is 32-bit on device;
  // go through size_t so the host pass accepts the truncation too).
  unsigned loff =
      (unsigned)(size_t)(__attribute__((address_space(3))) void*)lds_dst;
  unsigned long long g = (unsigned long long)gsrc;
  asm volatile("global_load_async_to_lds_b128 %0, %1, off"
               :: "v"(loff), "v"(g) : "memory");
}
__device__ inline void wait_async0() {
  asm volatile("s_wait_asynccnt 0x0" ::: "memory");
}

// ---------------------------------------------------------------------------
// fp32 -> bf16 conversion (x4 vectorized)
// ---------------------------------------------------------------------------
__global__ void cvt_f32_bf16_x4(const float* __restrict__ src,
                                __bf16* __restrict__ dst, long n4) {
  long i = (long)blockIdx.x * blockDim.x + threadIdx.x;
  if (i < n4) {
    float4 f = reinterpret_cast<const float4*>(src)[i];
    v4bf o;
    o[0] = (__bf16)f.x; o[1] = (__bf16)f.y;
    o[2] = (__bf16)f.z; o[3] = (__bf16)f.w;
    reinterpret_cast<v4bf*>(dst)[i] = o;
  }
}

// ---------------------------------------------------------------------------
// GEMM: C(MxN) = A(MxK) * B(KxN), A/B bf16 row-major, C = f32 or bf16.
// Block = 64x64 tile, 128 threads (4 waves), each wave a 32x32 quadrant
// (2x2 WMMA tiles). K stepped by 64 per LDS stage (2 WMMA K-steps, 8 WMMA
// per wave per barrier). A staged via async-to-LDS; B staged transposed
// through VGPRs ([n][k]) so B fragments are contiguous ds_load_b128 pairs.
// ---------------------------------------------------------------------------
template <typename OutT>
__global__ __launch_bounds__(128) void gemm_bf16_wmma(
    const __bf16* __restrict__ A, const __bf16* __restrict__ B,
    OutT* __restrict__ C, int M, int N, int K) {
  __shared__ __bf16 As[64][64];   // [m][k]   8 KB
  __shared__ __bf16 Bs[64][64];   // [n][k]   8 KB (transposed stage)

  const int tid  = threadIdx.x;
  const int lane = tid & 31;
  const int wid  = tid >> 5;
  const int bm   = blockIdx.x * 64;
  const int bn   = blockIdx.y * 64;
  const int wm   = (wid >> 1) * 32;      // wave row offset in tile
  const int wn   = (wid & 1) * 32;       // wave col offset in tile
  const int half = lane >> 4;
  const int l15  = lane & 15;
  const int kbase = half * 8;            // A-fragment K sub-offset
  const int khalf = half * 16;           // unused (B fragments use kbase)

  v8f acc[2][2] = {};

  for (int k0 = 0; k0 < K; k0 += 64) {
    // --- stage A tile 64x64 via async global->LDS (512 x b128, 4/thread) --
#pragma unroll
    for (int i = 0; i < 4; ++i) {
      int idx = i * 128 + tid;
      int r = idx >> 3, c8 = (idx & 7) * 8;
      async_copy_b128(&As[r][c8], A + (size_t)(bm + r) * K + (k0 + c8));
    }
    // --- stage B tile transposed: 64 k-rows x 64 n-cols -------------------
#pragma unroll
    for (int i = 0; i < 4; ++i) {
      int idx = i * 128 + tid;
      int kr = idx >> 3, n8 = (idx & 7) * 8;
      const __bf16* gp = B + (size_t)(k0 + kr) * N + (bn + n8);
      uint4 t = *reinterpret_cast<const uint4*>(gp);
      __builtin_prefetch(gp + (size_t)64 * N, 0, 1);   // next K stage
      const __bf16* e = reinterpret_cast<const __bf16*>(&t);
#pragma unroll
      for (int j = 0; j < 8; ++j) Bs[n8 + j][kr] = e[j];
    }
    wait_async0();
    __syncthreads();

    // --- fragments per documented CDNA5 VGPR layouts ----------------------
    v16bf af[2][2], bf[2][2];
#pragma unroll
    for (int mt = 0; mt < 2; ++mt)
#pragma unroll
      for (int kk = 0; kk < 2; ++kk) {
        const __bf16* rp = &As[wm + mt * 16 + l15][kk * 32];
        af[mt][kk] = frag2(rp + kbase, rp + 16 + kbase);
      }
#pragma unroll
    for (int nt = 0; nt < 2; ++nt)
#pragma unroll
      for (int kk = 0; kk < 2; ++kk) {
        const __bf16* rp = &Bs[wn + nt * 16 + l15][kk * 32];
        bf[nt][kk] = frag2(rp + kbase, rp + 16 + kbase);
      }
#pragma unroll
    for (int mt = 0; mt < 2; ++mt)
#pragma unroll
      for (int nt = 0; nt < 2; ++nt)
#pragma unroll
        for (int kk = 0; kk < 2; ++kk)
          acc[mt][nt] = wmma_bf16(af[mt][kk], bf[nt][kk], acc[mt][nt]);
    __syncthreads();
  }

  // --- store: lane holds col l15, rows half*8 + j ------------------------
#pragma unroll
  for (int mt = 0; mt < 2; ++mt)
#pragma unroll
    for (int nt = 0; nt < 2; ++nt) {
      int col = bn + wn + nt * 16 + l15;
#pragma unroll
      for (int j = 0; j < 8; ++j) {
        int row = bm + wm + mt * 16 + half * 8 + j;
        C[(size_t)row * N + col] = (OutT)acc[mt][nt][j];
      }
    }
}

// ---------------------------------------------------------------------------
// Flash attention per (b,h).  Q/K/V are bf16 in (B, L|R, H*DH) layout.
// Block: 64 query rows, 128 threads (4 waves).  Wave owns 16 query rows and
// all 64 head dims / key columns, so softmax row reductions stay inside one
// 16-lane half-wave.  Keys processed in chunks of 64 with online softmax.
// Q and K tiles staged via async global->LDS; V staged transposed.
// ---------------------------------------------------------------------------
__global__ __launch_bounds__(128) void flash_attn_bf16(
    const __bf16* __restrict__ Q, const __bf16* __restrict__ Kc,
    const __bf16* __restrict__ Vc, __bf16* __restrict__ O, int L, int R) {
  __shared__ __bf16 qs[64][64];        // [query][dh]
  __shared__ __bf16 kls[64][64];       // [key][dh]      (acts as Bt for S)
  __shared__ __bf16 vts[64][64];       // [dh][key]      (transposed V)
  __shared__ __bf16 ps[4][16][64];     // per-wave P tile [row][key]

  const int tid  = threadIdx.x;
  const int lane = tid & 31;
  const int wid  = tid >> 5;
  const int half = lane >> 4;
  const int l15  = lane & 15;
  const int bh = blockIdx.y;
  const int b  = bh >> 4;              // H == 16
  const int h  = bh & 15;
  const int q0 = blockIdx.x * 64;
  const int LD = 1024;                 // H * DH row stride

  const __bf16* qg = Q  + ((size_t)b * L + q0) * LD + h * 64;
  const __bf16* kg = Kc + (size_t)b * R * LD + h * 64;
  const __bf16* vg = Vc + (size_t)b * R * LD + h * 64;

  // stage Q tile (64x64) via async copy
#pragma unroll
  for (int i = 0; i < 4; ++i) {
    int idx = i * 128 + tid;
    int r = idx >> 3, c8 = (idx & 7) * 8;
    async_copy_b128(&qs[r][c8], qg + (size_t)r * LD + c8);
  }
  wait_async0();
  __syncthreads();

  // per-wave Q fragments (rows wid*16 + l15), two K-steps of 32
  v16bf qf[2];
#pragma unroll
  for (int kt = 0; kt < 2; ++kt) {
    const __bf16* rp = &qs[wid * 16 + l15][kt * 32];
    qf[kt] = frag2(rp + half * 8, rp + 16 + half * 8);
  }

  float row_m[8], row_l[8];
#pragma unroll
  for (int j = 0; j < 8; ++j) { row_m[j] = -1e30f; row_l[j] = 0.0f; }
  v8f oacc[4] = {};

  for (int r0 = 0; r0 < R; r0 += 64) {
    // stage K chunk row-major via async copy
#pragma unroll
    for (int i = 0; i < 4; ++i) {
      int idx = i * 128 + tid;
      int r = idx >> 3, c8 = (idx & 7) * 8;
      async_copy_b128(&kls[r][c8], kg + (size_t)(r0 + r) * LD + c8);
    }
    // stage V chunk transposed ([dh][key]) through VGPRs
#pragma unroll
    for (int i = 0; i < 4; ++i) {
      int idx = i * 128 + tid;
      int r = idx >> 3, c8 = (idx & 7) * 8;
      uint4 t = *reinterpret_cast<const uint4*>(vg + (size_t)(r0 + r) * LD + c8);
      const __bf16* e = reinterpret_cast<const __bf16*>(&t);
#pragma unroll
      for (int j = 0; j < 8; ++j) vts[c8 + j][r] = e[j];
    }
    wait_async0();
    __syncthreads();

    // ---- S = Q * K^T (per wave: 16 rows x 64 key cols) -------------------
    v8f s[4];
#pragma unroll
    for (int nt = 0; nt < 4; ++nt) {
      v8f c = {};
#pragma unroll
      for (int kt = 0; kt < 2; ++kt) {
        const __bf16* rp = &kls[nt * 16 + l15][kt * 32];
        c = wmma_bf16(qf[kt], frag2(rp + half * 8, rp + 16 + half * 8), c);
      }
      s[nt] = c;
    }

    // ---- online softmax --------------------------------------------------
#pragma unroll
    for (int j = 0; j < 8; ++j) {
      float mx = -1e30f;
#pragma unroll
      for (int nt = 0; nt < 4; ++nt) {
        s[nt][j] *= ATTN_SCALE;
        mx = fmaxf(mx, s[nt][j]);
      }
#pragma unroll
      for (int d = 1; d < 16; d <<= 1) mx = fmaxf(mx, __shfl_xor(mx, d, 32));
      float nm = fmaxf(row_m[j], mx);

      float sum = 0.0f;
#pragma unroll
      for (int nt = 0; nt < 4; ++nt) {
        float p = __expf(s[nt][j] - nm);
        s[nt][j] = p;
        sum += p;
      }
#pragma unroll
      for (int d = 1; d < 16; d <<= 1) sum += __shfl_xor(sum, d, 32);

      float alpha = __expf(row_m[j] - nm);
      row_l[j] = row_l[j] * alpha + sum;
      row_m[j] = nm;
#pragma unroll
      for (int nt = 0; nt < 4; ++nt) oacc[nt][j] *= alpha;
      // spill P row to per-wave LDS (D-layout -> memory)
#pragma unroll
      for (int nt = 0; nt < 4; ++nt)
        ps[wid][half * 8 + j][nt * 16 + l15] = (__bf16)s[nt][j];
    }

    // ---- O += P * V  (re-load P in A-layout) -----------------------------
    v16bf pf[2];
#pragma unroll
    for (int kt = 0; kt < 2; ++kt) {
      const __bf16* rp = &ps[wid][l15][kt * 32];
      pf[kt] = frag2(rp + half * 8, rp + 16 + half * 8);
    }
#pragma unroll
    for (int nt = 0; nt < 4; ++nt)
#pragma unroll
      for (int kt = 0; kt < 2; ++kt) {
        const __bf16* rp = &vts[nt * 16 + l15][kt * 32];
        oacc[nt] = wmma_bf16(pf[kt], frag2(rp + half * 8, rp + 16 + half * 8), oacc[nt]);
      }
    __syncthreads();
  }

  // ---- finalize: O /= l, store bf16 --------------------------------------
  __bf16* og = O + ((size_t)b * L + q0 + wid * 16) * LD + h * 64;
#pragma unroll
  for (int j = 0; j < 8; ++j) {
    float inv = 1.0f / row_l[j];
#pragma unroll
    for (int nt = 0; nt < 4; ++nt)
      og[(size_t)(half * 8 + j) * LD + nt * 16 + l15] =
          (__bf16)(oacc[nt][j] * inv);
  }
}

// ---------------------------------------------------------------------------
// Host orchestration
// ---------------------------------------------------------------------------
extern "C" void kernel_launch(void* const* d_in, const int* in_sizes, int n_in,
                              void* d_out, int out_size, void* d_ws,
                              size_t ws_size, hipStream_t stream) {
  const int B = 4, L = 4096, R = 512, DIM = 1024, INNER = 1024;
  const int ML = B * L;      // 16384 rows for x-side GEMMs
  const int MR = B * R;      // 2048 rows for context-side GEMMs

  const float* x   = (const float*)d_in[0];
  const float* ctx = (const float*)d_in[1];
  const float* Wq  = (const float*)d_in[2];
  const float* Wk  = (const float*)d_in[3];
  const float* Wv  = (const float*)d_in[4];
  const float* Wo  = (const float*)d_in[5];
  float* out = (float*)d_out;

  // workspace carve-up (bf16 buffers)
  char* w = (char*)d_ws;
  size_t o = 0;
  __bf16* xb  = (__bf16*)(w + o); o += (size_t)ML * DIM * 2;    // 32 MB
  __bf16* cb  = (__bf16*)(w + o); o += (size_t)MR * DIM * 2;    //  4 MB
  __bf16* wqb = (__bf16*)(w + o); o += (size_t)DIM * INNER * 2; //  2 MB
  __bf16* wkb = (__bf16*)(w + o); o += (size_t)DIM * INNER * 2;
  __bf16* wvb = (__bf16*)(w + o); o += (size_t)DIM * INNER * 2;
  __bf16* wob = (__bf16*)(w + o); o += (size_t)INNER * DIM * 2;
  __bf16* qb  = (__bf16*)(w + o); o += (size_t)ML * INNER * 2;  // 32 MB
  __bf16* kb  = (__bf16*)(w + o); o += (size_t)MR * INNER * 2;  //  4 MB
  __bf16* vb  = (__bf16*)(w + o); o += (size_t)MR * INNER * 2;  //  4 MB
  __bf16* ab  = (__bf16*)(w + o); o += (size_t)ML * INNER * 2;  // 32 MB

  // 1) fp32 -> bf16 conversions
  auto cvt = [&](const float* s, __bf16* d, long n) {
    long n4 = n / 4;
    cvt_f32_bf16_x4<<<(n4 + 255) / 256, 256, 0, stream>>>(s, d, n4);
  };
  cvt(x,   xb,  (long)ML * DIM);
  cvt(ctx, cb,  (long)MR * DIM);
  cvt(Wq,  wqb, (long)DIM * INNER);
  cvt(Wk,  wkb, (long)DIM * INNER);
  cvt(Wv,  wvb, (long)DIM * INNER);
  cvt(Wo,  wob, (long)INNER * DIM);

  // 2) projections: Q = x*Wq, K = ctx*Wk, V = ctx*Wv   (bf16 out)
  gemm_bf16_wmma<__bf16><<<dim3(ML / 64, INNER / 64), 128, 0, stream>>>(
      xb, wqb, qb, ML, INNER, DIM);
  gemm_bf16_wmma<__bf16><<<dim3(MR / 64, INNER / 64), 128, 0, stream>>>(
      cb, wkb, kb, MR, INNER, DIM);
  gemm_bf16_wmma<__bf16><<<dim3(MR / 64, INNER / 64), 128, 0, stream>>>(
      cb, wvb, vb, MR, INNER, DIM);

  // 3) attention: grid = (L/64 query tiles, B*H heads)
  flash_attn_bf16<<<dim3(L / 64, B * 16), 128, 0, stream>>>(
      qb, kb, vb, ab, L, R);

  // 4) output projection: out = attn * Wo  (f32 out)
  gemm_bf16_wmma<float><<<dim3(ML / 64, DIM / 64), 128, 0, stream>>>(
      ab, wob, out, ML, DIM, INNER);
}
